// MLA_76785425318111
// MI455X (gfx1250) — matrix operand order repaired and checked by reading, not compiled, not run on hardware
//
#include <hip/hip_runtime.h>
#include <hip/hip_bf16.h>
#include <math.h>

// ---------------------------------------------------------------------------
// MLA forward for MI455X (gfx1250, wave32, WMMA bf16).
// Compute-bound (~258 GFLOP vs ~150MB traffic @23.3TB/s) -> everything runs
// through v_wmma_f32_16x16x32_bf16 with on-the-fly fp32->bf16 conversion.
// K-tile staging in the attention kernel uses GLOBAL_LOAD_ASYNC_TO_LDS_B128
// (raw bf16 copy, tracked on ASYNCcnt).
// ---------------------------------------------------------------------------

typedef __attribute__((ext_vector_type(16))) __bf16 bf16x16;
typedef __attribute__((ext_vector_type(2)))  __bf16 bf16x2;
typedef __attribute__((ext_vector_type(8)))  float  f32x8;

#define B_    4
#define T_    1024
#define DIM_  2048
#define H_    16
#define NOPE_ 128
#define ROPE_ 64
#define HD_   192
#define VD_   128

// fp32 -> bf16 round-to-nearest-even, cheapest available path.
static __device__ __forceinline__ unsigned short f2bf(float f) {
  unsigned u = __float_as_uint(f);
  u += 0x7FFFu + ((u >> 16) & 1u);
  return (unsigned short)(u >> 16);
}
static __device__ __forceinline__ unsigned pack2(float a, float b) {
#if __has_builtin(__builtin_amdgcn_cvt_pk_bf16_f32)
  union { bf16x2 v; unsigned u; } c;
  c.v = __builtin_amdgcn_cvt_pk_bf16_f32(a, b);
  return c.u;
#else
  return (unsigned)f2bf(a) | ((unsigned)f2bf(b) << 16);
#endif
}

union Frag16 { uint4 u[2]; bf16x16 v; };

// Build a 16-element bf16 fragment from two 16B chunks in (LDS or global) mem.
static __device__ __forceinline__ bf16x16 load_frag(const unsigned short* p,
                                                    int o0, int o1) {
  Frag16 f;
  f.u[0] = *(const uint4*)(p + o0);
  f.u[1] = *(const uint4*)(p + o1);
  return f.v;
}

// Async raw copy: 16 bytes global -> LDS, per lane. GVS addressing:
// mem = saddr(64b SGPR) + vaddr(32b byte offset); LDS dest = vdst VGPR offset.
static __device__ __forceinline__ void async_copy_b128(unsigned lds_off,
                                                       unsigned gbyte_off,
                                                       const void* sbase) {
  asm volatile("global_load_async_to_lds_b128 %0, %1, %2"
               :
               : "v"(lds_off), "v"(gbyte_off), "s"(sbase)
               : "memory");
}
static __device__ __forceinline__ void wait_asynccnt0() {
  asm volatile("s_wait_asynccnt 0x0" ::: "memory");
}

// ---------------------------------------------------------------------------
// Tiled GEMM:  C[M,N] = A[M,K](fp32) * W[K,N](fp32) + bias, bf16 WMMA core.
// Block tile 128x128x32, 256 threads = 8 waves in a 2x4 grid,
// wave tile 64x32 = 4x2 C-tiles of 16x16.
// EPI: 0 = Q/K projection (RoPE + scale, bf16 out)
//      1 = V projection   (bf16 out)
//      2 = output proj    (fp32 out)
// ---------------------------------------------------------------------------
template <int EPI>
__global__ __launch_bounds__(256) void gemm_kernel(
    const float* __restrict__ A, const float* __restrict__ W,
    const float* __restrict__ bias, void* __restrict__ Cout,
    int M, int N, int K, const float* __restrict__ freqs, float oscale) {
  __shared__ unsigned short As[128 * 40];   // [row][k], stride 40 bf16
  __shared__ unsigned short Bs[128 * 40];   // [n][k] (transposed), stride 40

  const int tid  = threadIdx.x;
  const int lane = tid & 31;
  const int w    = tid >> 5;
  const int wr   = w >> 2;      // 0..1
  const int wc   = w & 3;       // 0..3
  const int m0   = blockIdx.y * 128;
  const int n0   = blockIdx.x * 128;
  const int lm   = lane & 15;
  const int hi   = lane >> 4;   // 0 or 1

  f32x8 acc[4][2];
  for (int i = 0; i < 4; ++i)
    for (int j = 0; j < 2; ++j)
      for (int e = 0; e < 8; ++e) acc[i][j][e] = 0.0f;

  // staging coordinates
  const int ar  = tid >> 3;          // A row 0..31 (+i*32)
  const int ak  = (tid & 7) * 4;     // A k    0..28
  const int bkr = tid >> 5;          // W row  0..7 (+i*8)
  const int bn  = (tid & 31) * 4;    // W col  0..124

  const int KT = K >> 5;
  float4 areg[4], breg[4];

  // prologue: load k-tile 0 into registers
  for (int i = 0; i < 4; ++i)
    areg[i] = *(const float4*)&A[(size_t)(m0 + ar + i * 32) * K + ak];
  for (int i = 0; i < 4; ++i)
    breg[i] = *(const float4*)&W[(size_t)(bkr + i * 8) * N + n0 + bn];

  for (int kt = 0; kt < KT; ++kt) {
    // store staged registers -> LDS (convert to bf16, packed pairs)
    for (int i = 0; i < 4; ++i) {
      unsigned short* d = &As[(ar + i * 32) * 40 + ak];
      *(unsigned*)(d + 0) = pack2(areg[i].x, areg[i].y);
      *(unsigned*)(d + 2) = pack2(areg[i].z, areg[i].w);
    }
    for (int i = 0; i < 4; ++i) {
      const int kr = bkr + i * 8;
      Bs[(bn + 0) * 40 + kr] = f2bf(breg[i].x);
      Bs[(bn + 1) * 40 + kr] = f2bf(breg[i].y);
      Bs[(bn + 2) * 40 + kr] = f2bf(breg[i].z);
      Bs[(bn + 3) * 40 + kr] = f2bf(breg[i].w);
    }
    __syncthreads();

    // prefetch next k-tile while we compute on LDS
    if (kt + 1 < KT) {
      const int k1 = (kt + 1) * 32;
      for (int i = 0; i < 4; ++i)
        areg[i] = *(const float4*)&A[(size_t)(m0 + ar + i * 32) * K + k1 + ak];
      for (int i = 0; i < 4; ++i)
        breg[i] = *(const float4*)&W[(size_t)(k1 + bkr + i * 8) * N + n0 + bn];
    }

    // fragments (ISA 7.12.2 16-bit A 16x32 / B 32x16 layouts)
    bf16x16 afr[4], bfr[2];
    const int ah = hi ? 8 : 0;
    const int bh = hi ? 8 : 0;
    for (int i = 0; i < 4; ++i) {
      const int base = (wr * 64 + i * 16 + lm) * 40 + ah;
      afr[i] = load_frag(As, base, base + 16);
    }
    for (int j = 0; j < 2; ++j) {
      const int base = (wc * 32 + j * 16 + lm) * 40 + bh;
      bfr[j] = load_frag(Bs, base, base + 16);
    }
    for (int i = 0; i < 4; ++i)
      for (int j = 0; j < 2; ++j)
        acc[i][j] = __builtin_amdgcn_wmma_f32_16x16x32_bf16(
            false, afr[i], false, bfr[j], (short)0, acc[i][j], false, false);
    __syncthreads();
  }

  // epilogue
  for (int i = 0; i < 4; ++i) {
    const int gmb = m0 + wr * 64 + i * 16 + hi * 8;
    for (int j = 0; j < 2; ++j) {
      const int gn = n0 + wc * 32 + j * 16 + lm;
      const float bv = bias[gn];
      if constexpr (EPI == 0) {
        const int d = gn % HD_;
        const bool isrope = (d >= NOPE_);     // uniform per 16-col tile
        const int p = (d - NOPE_) >> 1;
        const bool odd = (d & 1);
        for (int jj = 0; jj < 8; ++jj) {
          const int gm = gmb + jj;
          float val = acc[i][j][jj] + bv;
          if (isrope) {
            const int t = gm & (T_ - 1);
            const float c = freqs[t * ROPE_ + p * 2];
            const float s = freqs[t * ROPE_ + p * 2 + 1];
            const float pv = __shfl_xor(val, 1, 32);  // partner column
            val = odd ? (val * c + pv * s) : (val * c - pv * s);
          }
          val *= oscale;
          ((unsigned short*)Cout)[(size_t)gm * N + gn] = f2bf(val);
        }
      } else if constexpr (EPI == 1) {
        for (int jj = 0; jj < 8; ++jj) {
          const int gm = gmb + jj;
          ((unsigned short*)Cout)[(size_t)gm * N + gn] = f2bf(acc[i][j][jj] + bv);
        }
      } else {
        for (int jj = 0; jj < 8; ++jj) {
          const int gm = gmb + jj;
          ((float*)Cout)[(size_t)gm * N + gn] = acc[i][j][jj] + bv;
        }
      }
    }
  }
}

// ---------------------------------------------------------------------------
// Flash attention. Grid (T/128, H, B), 256 threads = 8 waves,
// one wave = 16 q rows. q is pre-scaled by 1/sqrt(HD); causal mask inline.
// K staging uses async global->LDS copies (ASYNCcnt).
// ---------------------------------------------------------------------------
__global__ __launch_bounds__(256) void attn_kernel(
    const unsigned short* __restrict__ Q, const unsigned short* __restrict__ Kb,
    const unsigned short* __restrict__ Vb, float* __restrict__ Ctx) {
  __shared__ unsigned short Ks[32 * 200];     // [key][d]   (32x192 + pad)
  __shared__ unsigned short Vs[128 * 40];     // [vd][key]  (transposed)
  __shared__ unsigned short Ps[8][16 * 40];   // per-wave P scratch

  const int tid  = threadIdx.x;
  const int lane = tid & 31;
  const int w    = tid >> 5;
  const int qb   = blockIdx.x, h = blockIdx.y, b = blockIdx.z;
  const int t0   = qb * 128 + w * 16;
  const int lm   = lane & 15;
  const int hi   = lane >> 4;

  // Q fragments for this wave's 16 rows (6 chunks of K=32)
  bf16x16 qf[6];
  {
    const unsigned short* qrow =
        Q + (size_t)(b * T_ + t0 + lm) * (H_ * HD_) + h * HD_;
    const int ah = hi ? 8 : 0;
    for (int c = 0; c < 6; ++c) {
      Frag16 f;
      f.u[0] = *(const uint4*)(qrow + c * 32 + ah);
      f.u[1] = *(const uint4*)(qrow + c * 32 + ah + 16);
      qf[c] = f.v;
    }
  }

  f32x8 o[8];
  for (int i = 0; i < 8; ++i)
    for (int e = 0; e < 8; ++e) o[i][e] = 0.0f;
  float mrow[8], lrow[8];
  for (int jj = 0; jj < 8; ++jj) { mrow[jj] = -3.0e38f; lrow[jj] = 0.0f; }

  // uniform 64-bit bases for async staging (per (b,h))
  const unsigned short* kptr = Kb + (size_t)(b * T_) * (H_ * HD_) + h * HD_;
  const size_t vbase = (size_t)(b * T_) * (H_ * VD_) + h * VD_;

  const int nkb = qb * 4 + 4;                  // uniform over the workgroup
  for (int kb = 0; kb < nkb; ++kb) {
    __syncthreads();
    // stage K block via async DMA: 32 keys x 192 bf16 = 768 16B chunks
    for (int i = 0; i < 3; ++i) {
      const int c = tid + i * 256;
      const int r = c / 24, col = (c % 24) * 8;
      const unsigned lds = (unsigned)(size_t)(const void*)&Ks[r * 200 + col];
      const unsigned gof = (unsigned)(((kb * 32 + r) * (H_ * HD_) + col) * 2);
      async_copy_b128(lds, gof, (const void*)kptr);
    }
    // stage V block transposed (needs element shuffle -> manual): 32 x 128
    for (int i = 0; i < 2; ++i) {
      const int c = tid + i * 256;
      const int r = c >> 4, col = (c & 15) * 8;
      uint4 dv = *(const uint4*)(Vb + vbase + (size_t)(kb * 32 + r) * (H_ * VD_) + col);
      const unsigned short* e = (const unsigned short*)&dv;
      for (int q = 0; q < 8; ++q) Vs[(col + q) * 40 + r] = e[q];
    }
    wait_asynccnt0();        // this wave's async K copies landed in LDS
    __syncthreads();         // + everyone else's

    // S = Q K^T : two 16-key tiles
    f32x8 sa[2];
    for (int j = 0; j < 2; ++j) {
      for (int e = 0; e < 8; ++e) sa[j][e] = 0.0f;
      const int key = j * 16 + lm;
      for (int c = 0; c < 6; ++c) {
        const int base = key * 200 + c * 32 + (hi ? 8 : 0);
        bf16x16 kf = load_frag(Ks, base, base + 16);
        sa[j] = __builtin_amdgcn_wmma_f32_16x16x32_bf16(
            false, qf[c], false, kf, (short)0, sa[j], false, false);
      }
    }

    // causal mask + online softmax
    float s0[8], s1[8], corr[8];
    const int sg0 = kb * 32 + lm, sg1 = sg0 + 16;
    for (int jj = 0; jj < 8; ++jj) {
      const int tg = t0 + jj + hi * 8;
      s0[jj] = (sg0 <= tg) ? sa[0][jj] : -1.0e9f;
      s1[jj] = (sg1 <= tg) ? sa[1][jj] : -1.0e9f;
    }
    for (int jj = 0; jj < 8; ++jj) {
      float mb = fmaxf(s0[jj], s1[jj]);
      for (int off = 1; off < 16; off <<= 1)
        mb = fmaxf(mb, __shfl_xor(mb, off, 32));
      const float mn = fmaxf(mrow[jj], mb);
      corr[jj] = __expf(mrow[jj] - mn);
      mrow[jj] = mn;
      s0[jj] = __expf(s0[jj] - mn);
      s1[jj] = __expf(s1[jj] - mn);
      float rs = s0[jj] + s1[jj];
      for (int off = 1; off < 16; off <<= 1) rs += __shfl_xor(rs, off, 32);
      lrow[jj] = lrow[jj] * corr[jj] + rs;
    }
    for (int i = 0; i < 8; ++i)
      for (int jj = 0; jj < 8; ++jj) o[i][jj] *= corr[jj];

    // re-shape P (C layout) into an A fragment via per-wave LDS scratch
    unsigned short* P = &Ps[w][0];
    for (int jj = 0; jj < 8; ++jj) {
      const int r = jj + hi * 8;
      P[r * 40 + lm]      = f2bf(s0[jj]);
      P[r * 40 + 16 + lm] = f2bf(s1[jj]);
    }
    const int pb = lm * 40 + (hi ? 8 : 0);
    bf16x16 pf = load_frag(P, pb, pb + 16);

    // O += P V
    for (int tv = 0; tv < 8; ++tv) {
      const int base = (tv * 16 + lm) * 40 + (hi ? 8 : 0);
      bf16x16 vf = load_frag(Vs, base, base + 16);
      o[tv] = __builtin_amdgcn_wmma_f32_16x16x32_bf16(
          false, pf, false, vf, (short)0, o[tv], false, false);
    }
  }

  // normalize and write ctx (fp32, layout (B,T,H*VD))
  float inv[8];
  for (int jj = 0; jj < 8; ++jj) inv[jj] = 1.0f / lrow[jj];
  for (int tv = 0; tv < 8; ++tv) {
    const int vd = tv * 16 + lm;
    for (int jj = 0; jj < 8; ++jj) {
      const int t = t0 + jj + hi * 8;
      Ctx[(size_t)(b * T_ + t) * (H_ * VD_) + h * VD_ + vd] = o[tv][jj] * inv[jj];
    }
  }
}

// ---------------------------------------------------------------------------
extern "C" void kernel_launch(void* const* d_in, const int* in_sizes, int n_in,
                              void* d_out, int out_size, void* d_ws, size_t ws_size,
                              hipStream_t stream) {
  (void)in_sizes; (void)n_in; (void)out_size; (void)ws_size;
  const float* x  = (const float*)d_in[0];
  const float* fc = (const float*)d_in[1];   // freqs_cis (T,1,ROPE/2,2)
  // d_in[2] = mask: causal mask is computed inline, input unused
  const float* Wq = (const float*)d_in[3];
  const float* bq = (const float*)d_in[4];
  const float* Wk = (const float*)d_in[5];
  const float* bk = (const float*)d_in[6];
  const float* Wv = (const float*)d_in[7];
  const float* bv = (const float*)d_in[8];
  const float* Wo = (const float*)d_in[9];
  const float* bo = (const float*)d_in[10];
  float* out = (float*)d_out;

  char* ws = (char*)d_ws;
  unsigned short* qbuf = (unsigned short*)(ws + 0);          // 4096*3072 bf16
  unsigned short* kbuf = (unsigned short*)(ws + 25165824);   // 4096*3072 bf16
  unsigned short* vbuf = (unsigned short*)(ws + 50331648);   // 4096*2048 bf16
  float*          ctx  = (float*)(ws + 67108864);            // 4096*2048 fp32

  const int M = B_ * T_;
  const dim3 blk(256);
  const float qscale = 1.0f / sqrtf((float)HD_);

  // Q projection (+bias, RoPE, 1/sqrt(HD) folded in) -> bf16
  gemm_kernel<0><<<dim3((H_ * HD_) / 128, M / 128), blk, 0, stream>>>(
      x, Wq, bq, qbuf, M, H_ * HD_, DIM_, fc, qscale);
  // K projection (+bias, RoPE) -> bf16
  gemm_kernel<0><<<dim3((H_ * HD_) / 128, M / 128), blk, 0, stream>>>(
      x, Wk, bk, kbuf, M, H_ * HD_, DIM_, fc, 1.0f);
  // V projection (+bias) -> bf16
  gemm_kernel<1><<<dim3((H_ * VD_) / 128, M / 128), blk, 0, stream>>>(
      x, Wv, bv, vbuf, M, H_ * VD_, DIM_, fc, 1.0f);
  // flash attention -> ctx fp32
  attn_kernel<<<dim3(T_ / 128, H_, B_), blk, 0, stream>>>(qbuf, kbuf, vbuf, ctx);
  // output projection -> fp32 d_out
  gemm_kernel<2><<<dim3(DIM_ / 128, M / 128), blk, 0, stream>>>(
      ctx, Wo, bo, out, M, DIM_, DIM_, fc, 1.0f);
}
